// EdgeProcessor_47768626266213
// MI455X (gfx1250) — matrix operand order repaired and checked by reading, not compiled, not run on hardware
//
#include <hip/hip_runtime.h>

typedef __attribute__((ext_vector_type(16))) __bf16 v16bf;
typedef __attribute__((ext_vector_type(8)))  __bf16 v8bf;
typedef __attribute__((ext_vector_type(4)))  __bf16 v4bf;
typedef __attribute__((ext_vector_type(8)))  float  v8f;

#define CAT    384
#define LAT    128
#define WAVES  8
#define TILE_M 16
#define BLK_E  (WAVES * TILE_M)   // 128 edges per block

// LDS layout (offsets in __bf16 elements)
#define W0_OFF   0                         // W0^T  [128][384]
#define W1_OFF   (LAT * CAT)               // W1^T  [128][128]
#define W2_OFF   (W1_OFF + LAT * LAT)      // W2^T  [128][128]
#define ACT_OFF  (W2_OFF + LAT * LAT)      // per-wave [16][384] bf16 tiles
#define ACT_PER_WAVE (TILE_M * CAT)
#define LDS_BYTES ((ACT_OFF + WAVES * ACT_PER_WAVE) * 2)   // 262144 B

// ---- WMMA fragment loaders (wave32; A and B both use contiguous k = 16*(l>>4) + s) ----
// A 16x32 bf16: lane l holds row m=l&15; K chunks [g*16 .. g*16+7] and [g*16+8 ..]
// where g = l>>4 -> two contiguous 16B LDS loads.
static __device__ inline v16bf load_a_frag(const __bf16* base, int ld, int kstep,
                                           int ln, int hi) {
    const __bf16* p1 = base + ln * ld + kstep * 32 + hi * 16;
    const __bf16* p2 = p1 + 8;
    union { v16bf v; v8bf h[2]; } u;
    u.h[0] = *(const v8bf*)p1;
    u.h[1] = *(const v8bf*)p2;
    return u.v;
}
// B 32x16 bf16: lane l holds col n=l&15; K = (l>>4)*16 + 2v+{0,1}
// -> 16 consecutive K values = 32B contiguous in the [n][k] transposed layout.
static __device__ inline v16bf load_b_frag(const __bf16* wT, int ld, int nt,
                                           int kstep, int ln, int hi) {
    const __bf16* p = wT + (nt * 16 + ln) * ld + kstep * 32 + hi * 16;
    return *(const v16bf*)p;
}

// Pipelined 16xK @ Kx128 GEMM: batch the 8 B fragments of each k-step into a
// register array and prefetch the next A fragment, so the 8-WMMA chain overlaps
// the outstanding DS loads (staggered s_wait_dscnt instead of dscnt==0 stalls).
template <int KSTEPS>
static __device__ inline void gemm_tile(const __bf16* aBase, int aLd,
                                        const __bf16* wT, int wLd,
                                        v8f acc[8], int ln, int hi) {
    v16bf a_cur = load_a_frag(aBase, aLd, 0, ln, hi);
    #pragma unroll
    for (int ks = 0; ks < KSTEPS; ++ks) {
        v16bf b[8];
        #pragma unroll
        for (int nt = 0; nt < 8; ++nt)
            b[nt] = load_b_frag(wT, wLd, nt, ks, ln, hi);
        v16bf a_next = a_cur;
        if (ks + 1 < KSTEPS)
            a_next = load_a_frag(aBase, aLd, ks + 1, ln, hi);
        #pragma unroll
        for (int nt = 0; nt < 8; ++nt)
            acc[nt] = __builtin_amdgcn_wmma_f32_16x16x32_bf16(
                false, a_cur, false, b[nt], (short)0, acc[nt], false, false);
        a_cur = a_next;
    }
}

extern "C" __global__ __launch_bounds__(256)
void edge_mlp_wmma(const float* __restrict__ sf, const float* __restrict__ rf,
                   const float* __restrict__ ef, const int* __restrict__ snd,
                   const int* __restrict__ rcv,
                   const float* __restrict__ W0, const float* __restrict__ b0,
                   const float* __restrict__ W1, const float* __restrict__ b1,
                   const float* __restrict__ W2, const float* __restrict__ b2,
                   const float* __restrict__ gamma, const float* __restrict__ beta,
                   float* __restrict__ out, int E)
{
    extern __shared__ char smem_raw[];
    __bf16* lds = (__bf16*)smem_raw;

    const int tid  = threadIdx.x;
    const int wave = tid >> 5;
    const int lane = tid & 31;
    const int ln   = lane & 15;
    const int hi   = lane >> 4;

    // ---------- stage weights into LDS, transposed [n][k], bf16 ----------
    for (int idx = tid; idx < LAT * (CAT / 2); idx += 256) {
        int n  = idx / (CAT / 2);
        int k2 = (idx % (CAT / 2)) * 2;
        union { struct { __bf16 a, b; } s; unsigned u; } p;
        p.s.a = (__bf16)W0[(k2 + 0) * LAT + n];
        p.s.b = (__bf16)W0[(k2 + 1) * LAT + n];
        *(unsigned*)(lds + W0_OFF + n * CAT + k2) = p.u;
    }
    for (int idx = tid; idx < LAT * (LAT / 2); idx += 256) {
        int n  = idx / (LAT / 2);
        int k2 = (idx % (LAT / 2)) * 2;
        {
            union { struct { __bf16 a, b; } s; unsigned u; } p;
            p.s.a = (__bf16)W1[(k2 + 0) * LAT + n];
            p.s.b = (__bf16)W1[(k2 + 1) * LAT + n];
            *(unsigned*)(lds + W1_OFF + n * LAT + k2) = p.u;
        }
        {
            union { struct { __bf16 a, b; } s; unsigned u; } p;
            p.s.a = (__bf16)W2[(k2 + 0) * LAT + n];
            p.s.b = (__bf16)W2[(k2 + 1) * LAT + n];
            *(unsigned*)(lds + W2_OFF + n * LAT + k2) = p.u;
        }
    }

    // ---------- gather + stage this wave's 16 edges: [16][384] bf16 ----------
    const long eBase = (long)blockIdx.x * BLK_E + wave * TILE_M;
    __bf16* act = lds + ACT_OFF + wave * ACT_PER_WAVE;
    #pragma unroll 4
    for (int e = 0; e < TILE_M; ++e) {
        long eid = eBase + e;
        long eu  = (eid < E) ? eid : (long)(E - 1);   // wave-uniform -> s_load
        int  s   = snd[eu];
        int  r   = rcv[eu];
        float4 vs = ((const float4*)(sf + (long)s * LAT))[lane];
        float4 vr = ((const float4*)(rf + (long)r * LAT))[lane];
        float4 ve = ((const float4*)(ef + eu * LAT))[lane];
        __bf16* d = act + e * CAT;
        int j = lane * 4;
        { v4bf p; p.x=(__bf16)vs.x; p.y=(__bf16)vs.y; p.z=(__bf16)vs.z; p.w=(__bf16)vs.w;
          *(v4bf*)(d + j) = p; }
        { v4bf p; p.x=(__bf16)vr.x; p.y=(__bf16)vr.y; p.z=(__bf16)vr.z; p.w=(__bf16)vr.w;
          *(v4bf*)(d + 128 + j) = p; }
        { v4bf p; p.x=(__bf16)ve.x; p.y=(__bf16)ve.y; p.z=(__bf16)ve.z; p.w=(__bf16)ve.w;
          *(v4bf*)(d + 256 + j) = p; }
    }
    __syncthreads();

    // ---------- layer 0: [16x384] @ [384x128] ----------
    v8f acc[8] = {};
    gemm_tile<CAT / 32>(act, CAT, lds + W0_OFF, CAT, acc, ln, hi);

    float bias0[8];
    #pragma unroll
    for (int nt = 0; nt < 8; ++nt) bias0[nt] = b0[nt * 16 + ln];
    #pragma unroll
    for (int nt = 0; nt < 8; ++nt)
        #pragma unroll
        for (int c = 0; c < 8; ++c) {
            float v = acc[nt][c] + bias0[nt];
            v = v > 0.0f ? v : 0.0f;
            act[(hi * 8 + c) * LAT + nt * 16 + ln] = (__bf16)v;   // DS in-order per wave
        }

    // ---------- layer 1: [16x128] @ [128x128] ----------
    v8f acc2[8] = {};
    gemm_tile<LAT / 32>(act, LAT, lds + W1_OFF, LAT, acc2, ln, hi);

    float bias1[8];
    #pragma unroll
    for (int nt = 0; nt < 8; ++nt) bias1[nt] = b1[nt * 16 + ln];
    #pragma unroll
    for (int nt = 0; nt < 8; ++nt)
        #pragma unroll
        for (int c = 0; c < 8; ++c) {
            float v = acc2[nt][c] + bias1[nt];
            v = v > 0.0f ? v : 0.0f;
            act[(hi * 8 + c) * LAT + nt * 16 + ln] = (__bf16)v;
        }

    // ---------- layer 2 (linear): [16x128] @ [128x128] ----------
    v8f acc3[8] = {};
    gemm_tile<LAT / 32>(act, LAT, lds + W2_OFF, LAT, acc3, ln, hi);

    // ---------- bias + LayerNorm + store ----------
    float bias2[8], gg[8], bb[8];
    #pragma unroll
    for (int nt = 0; nt < 8; ++nt) {
        int n = nt * 16 + ln;
        bias2[nt] = b2[n]; gg[nt] = gamma[n]; bb[nt] = beta[n];
    }
    // Row m = hi*8 + c lives in one 16-lane half; columns n = nt*16 + ln.
    #pragma unroll
    for (int c = 0; c < 8; ++c) {
        float sum = 0.0f, sq = 0.0f;
        float hv[8];
        #pragma unroll
        for (int nt = 0; nt < 8; ++nt) {
            float v = acc3[nt][c] + bias2[nt];
            hv[nt] = v;
            sum += v; sq += v * v;
        }
        // reduce across the 16 lanes of this half (xor 1,2,4,8 stays in-half)
        #pragma unroll
        for (int off = 1; off < 16; off <<= 1) {
            sum += __shfl_xor(sum, off, 32);
            sq  += __shfl_xor(sq,  off, 32);
        }
        float mu  = sum * (1.0f / 128.0f);
        float var = sq * (1.0f / 128.0f) - mu * mu;
        float rs  = __frsqrt_rn(var + 1e-3f);

        long row = eBase + hi * 8 + c;
        if (row < E) {
            float* orow = out + row * (long)LAT;
            #pragma unroll
            for (int nt = 0; nt < 8; ++nt)
                orow[nt * 16 + ln] = (hv[nt] - mu) * rs * gg[nt] + bb[nt];
        }
    }
}

extern "C" void kernel_launch(void* const* d_in, const int* in_sizes, int n_in,
                              void* d_out, int out_size, void* d_ws, size_t ws_size,
                              hipStream_t stream) {
    const float* sf = (const float*)d_in[0];
    const float* rf = (const float*)d_in[1];
    const float* ef = (const float*)d_in[2];
    const int*  snd = (const int*)d_in[3];
    const int*  rcv = (const int*)d_in[4];
    const float* W0 = (const float*)d_in[5];
    const float* b0 = (const float*)d_in[6];
    const float* W1 = (const float*)d_in[7];
    const float* b1 = (const float*)d_in[8];
    const float* W2 = (const float*)d_in[9];
    const float* b2 = (const float*)d_in[10];
    const float* gm = (const float*)d_in[11];
    const float* bt = (const float*)d_in[12];
    float* out = (float*)d_out;

    int E = in_sizes[3];                       // number of edges
    int grid = (E + BLK_E - 1) / BLK_E;

    // allow >64KB dynamic LDS (idempotent, not a stream op)
    hipFuncSetAttribute((const void*)edge_mlp_wmma,
                        hipFuncAttributeMaxDynamicSharedMemorySize, LDS_BYTES);

    edge_mlp_wmma<<<grid, 256, LDS_BYTES, stream>>>(
        sf, rf, ef, snd, rcv, W0, b0, W1, b1, W2, b2, gm, bt, out, E);
}
